// GraphSAGE_89034672046784
// MI455X (gfx1250) — compile-verified, matrix operand order checked by reading, not run on hardware
//
#include <hip/hip_runtime.h>
#include <hip/hip_bf16.h>
#include <math.h>

#define N_NODES 100000
#define N_EDGES 1600000
#define HID 128
#define MTILES (N_NODES / 16)   // 6250, exact

typedef __attribute__((ext_vector_type(16))) _Float16 v16h;
typedef __attribute__((ext_vector_type(8)))  float    v8f;

// ---------------------------------------------------------------------------
// A-operand loader: 16x32 f16 tile from fp32 row-major (ld=HID) activations.
// CDNA5 16-bit A layout: lanes 0-15 hold row (lane), K = {0..7, 16..23};
// lanes 16-31 hold row (lane-16), K = {8..15, 24..31}. v16h elems 0..7 = first
// group of 8 contiguous K, elems 8..15 = group at +16.
// `lanebase` = A + row*HID + g0 (per-lane, loop-invariant); kc is a constant.
// ---------------------------------------------------------------------------
template<bool CLAMP_NONFINITE>
__device__ __forceinline__ v16h load_a_f32(const float* __restrict__ lanebase, int kc) {
    const float* p = lanebase + (kc << 5);
    v16h a;
#pragma unroll
    for (int i = 0; i < 8; ++i) {
        float v0 = p[i];
        float v1 = p[i + 16];
        if (CLAMP_NONFINITE) {                    // segment_max of empty segment = -inf -> 0
            v0 = __builtin_isfinite(v0) ? v0 : 0.0f;
            v1 = __builtin_isfinite(v1) ? v1 : 0.0f;
        }
        a[i]     = (_Float16)v0;
        a[8 + i] = (_Float16)v1;
    }
    return a;
}

// B-operand loader from f16 weights, row-major [out][in] (computes act @ W^T).
// `lanebase` = W + (lane&15)*HID + g0 (per-lane, loop-invariant); nt/kc const.
__device__ __forceinline__ v16h load_b_f16(const _Float16* __restrict__ lanebase, int nt, int kc) {
    const _Float16* p = lanebase + nt * (16 * HID) + (kc << 5);
    v16h b;
#pragma unroll
    for (int i = 0; i < 8; ++i) { b[i] = p[i]; b[8 + i] = p[i + 16]; }
    return b;
}

// ---------------------------------------------------------------------------
// WMMA GEMM over nodes: out = A1 @ W1^T (+ A2 @ W2^T) + bias, optional lrelu,
// optional batchnorm sum/sumsq accumulation. 4 waves/block, 16 rows/wave.
// LDO is compile-time so every store folds its row stride into the 24-bit
// instruction offset (one address computation for the whole epilogue).
// ---------------------------------------------------------------------------
template<bool DUAL, bool STATS, int NT, bool LRELU, int LDO>
__global__ __launch_bounds__(128) void gemm_kernel(
    const float* __restrict__ A1, const float* __restrict__ A2,
    const _Float16* __restrict__ W1, const _Float16* __restrict__ W2,
    const float* __restrict__ bias, float* __restrict__ out,
    float* __restrict__ stats)
{
    const int lane  = threadIdx.x & 31;
    const int wave  = threadIdx.x >> 5;
    const int mtile = blockIdx.x * 4 + wave;
    if (mtile >= MTILES) return;                 // whole-wave uniform exit; EXEC stays full

    const int row16 = lane & 15;
    const int g0    = (lane >> 4) << 3;          // 0 or 8

    // per-lane loop-invariant base pointers
    const float*    a1b = A1 + (size_t)mtile * 16 * HID + row16 * HID + g0;
    const float*    a2b = DUAL ? (A2 + (size_t)mtile * 16 * HID + row16 * HID + g0) : nullptr;
    const _Float16* w1b = W1 + row16 * HID + g0;
    const _Float16* w2b = DUAL ? (W2 + row16 * HID + g0) : nullptr;

    v16h a1[4], a2[4];
#pragma unroll
    for (int c = 0; c < 4; ++c) a1[c] = load_a_f32<DUAL>(a1b, c);   // clamp only for agg
    if (DUAL) {
#pragma unroll
        for (int c = 0; c < 4; ++c) a2[c] = load_a_f32<false>(a2b, c);
    }

    const int r0 = mtile * 16 + g0;              // C/D layout: lane>=16 -> rows +8
    float* outb = out + (size_t)r0 * LDO + row16;     // all stores: immediate offsets
    const float* biasb = bias + row16;

#pragma unroll
    for (int nt = 0; nt < NT; ++nt) {
        // batch all B loads for this tile first -> one load clause, partial waits
        v16h b1[4], b2[4];
#pragma unroll
        for (int c = 0; c < 4; ++c) b1[c] = load_b_f16(w1b, nt, c);
        if (DUAL) {
#pragma unroll
            for (int c = 0; c < 4; ++c) b2[c] = load_b_f16(w2b, nt, c);
        }

        v8f acc = {};
#pragma unroll
        for (int c = 0; c < 4; ++c)
            acc = __builtin_amdgcn_wmma_f32_16x16x32_f16(false, a1[c], false, b1[c],
                                                         (short)0, acc, false, false);
        if (DUAL) {
#pragma unroll
            for (int c = 0; c < 4; ++c)
                acc = __builtin_amdgcn_wmma_f32_16x16x32_f16(false, a2[c], false, b2[c],
                                                             (short)0, acc, false, false);
        }

        const float bv = biasb[nt * 16];
        float s = 0.0f, s2 = 0.0f;
#pragma unroll
        for (int v = 0; v < 8; ++v) {
            float o = acc[v] + bv;
            if (LRELU) o = (o > 0.0f) ? o : 0.01f * o;
            outb[v * LDO + nt * 16] = o;         // immediate-offset global_store_b32
            if (STATS) { s += o; s2 += o * o; }
        }
        if (STATS) {
            // lanes L and L+16 hold the same column -> combine, then 16 atomics
            s  += __shfl_xor(s, 16, 32);
            s2 += __shfl_xor(s2, 16, 32);
            if (lane < 16) {
                atomicAdd(&stats[nt * 16 + row16], s);
                atomicAdd(&stats[HID + nt * 16 + row16], s2);
            }
        }
    }
}

// ---------------------------------------------------------------------------
// Edge scatter-max: one wave per edge, each lane handles 4 contiguous features.
// agg pre-filled with -inf; atomicMax(float) -> native fp32 atomic max in L2.
// ---------------------------------------------------------------------------
__global__ __launch_bounds__(256) void scatter_max_kernel(
    const float* __restrict__ h, const int* __restrict__ src,
    const int* __restrict__ dst, float* __restrict__ agg)
{
    const int gid = blockIdx.x * 256 + threadIdx.x;
    const int e = gid >> 5;
    if (e >= N_EDGES) return;
    const int f = (gid & 31) << 2;
    const int s = src[e];
    const int d = dst[e];
    const float4 v = *reinterpret_cast<const float4*>(h + (size_t)s * HID + f);
    float* q = agg + (size_t)d * HID + f;
    atomicMax(q + 0, v.x);
    atomicMax(q + 1, v.y);
    atomicMax(q + 2, v.z);
    atomicMax(q + 3, v.w);
}

__global__ __launch_bounds__(256) void init_kernel(float4* __restrict__ agg,
                                                   float* __restrict__ stats)
{
    const int i = blockIdx.x * 256 + threadIdx.x;
    const float ninf = -__builtin_huge_valf();
    if (i < N_NODES * HID / 4) agg[i] = make_float4(ninf, ninf, ninf, ninf);
    if (i < 2 * HID) stats[i] = 0.0f;
}

__global__ void bn_finalize_kernel(float* __restrict__ stats,
                                   const float* __restrict__ g,
                                   const float* __restrict__ be)
{
    const int c = threadIdx.x;
    if (c >= HID) return;
    const float inv_n = 1.0f / (float)N_NODES;
    const float mean = stats[c] * inv_n;
    const float var  = stats[HID + c] * inv_n - mean * mean;
    const float sc   = g[c] * rsqrtf(var + 1e-5f);
    stats[2 * HID + c] = sc;
    stats[3 * HID + c] = be[c] - mean * sc;
}

template<bool RES>
__global__ __launch_bounds__(256) void bn_lrelu_kernel(
    float4* __restrict__ buf, const float* __restrict__ stats,
    const float4* __restrict__ res)
{
    const int i = blockIdx.x * 256 + threadIdx.x;
    if (i >= N_NODES * HID / 4) return;
    const int c4 = (i & 31) << 2;
    const float4 v = buf[i];
    const float* sc = stats + 2 * HID + c4;
    const float* sh = stats + 3 * HID + c4;
    float o[4] = {v.x, v.y, v.z, v.w};
#pragma unroll
    for (int j = 0; j < 4; ++j) {
        float y = sc[j] * o[j] + sh[j];
        o[j] = (y > 0.0f) ? y : 0.01f * y;
    }
    if (RES) {
        const float4 rv = res[i];
        o[0] += rv.x; o[1] += rv.y; o[2] += rv.z; o[3] += rv.w;
    }
    buf[i] = make_float4(o[0], o[1], o[2], o[3]);
}

__global__ void cvt_f16_kernel(const float* __restrict__ w, _Float16* __restrict__ o, int n) {
    const int i = blockIdx.x * 256 + threadIdx.x;
    if (i < n) o[i] = (_Float16)w[i];
}

__global__ __launch_bounds__(256) void fc2_kernel(
    const float* __restrict__ h1, const float* __restrict__ w,
    const float* __restrict__ b, float* __restrict__ out)
{
    const int i = blockIdx.x * 256 + threadIdx.x;
    if (i >= N_NODES) return;
    const float4* hp = reinterpret_cast<const float4*>(h1 + (size_t)i * 64);
    const float4* wp = reinterpret_cast<const float4*>(w);
    float acc = 0.0f;
#pragma unroll
    for (int j = 0; j < 16; ++j) {
        const float4 a = hp[j], ww = wp[j];
        acc += a.x * ww.x + a.y * ww.y + a.z * ww.z + a.w * ww.w;
    }
    out[i] = acc + b[0];
}

// ---------------------------------------------------------------------------
extern "C" void kernel_launch(void* const* d_in, const int* in_sizes, int n_in,
                              void* d_out, int out_size, void* d_ws, size_t ws_size,
                              hipStream_t stream)
{
    const float* x    = (const float*)d_in[0];
    const int*   ei   = (const int*)d_in[1];
    const int*   src  = ei;
    const int*   dst  = ei + N_EDGES;
    const float* W_in = (const float*)d_in[2];
    const float* b_in = (const float*)d_in[3];
    const float* W_l[3] = {(const float*)d_in[4], (const float*)d_in[7], (const float*)d_in[10]};
    const float* b_l[3] = {(const float*)d_in[5], (const float*)d_in[8], (const float*)d_in[11]};
    const float* W_r[3] = {(const float*)d_in[6], (const float*)d_in[9], (const float*)d_in[12]};
    const float* g[3]   = {(const float*)d_in[13], (const float*)d_in[15], (const float*)d_in[17]};
    const float* be[3]  = {(const float*)d_in[14], (const float*)d_in[16], (const float*)d_in[18]};
    const float* W_fc1  = (const float*)d_in[19];
    const float* b_fc1  = (const float*)d_in[20];
    const float* W_fc2  = (const float*)d_in[21];
    const float* b_fc2  = (const float*)d_in[22];

    // Workspace layout (floats): h0 | bufA | bufB | stats(512) | f16 weights
    float* ws = (float*)d_ws;
    const size_t NF = (size_t)N_NODES * HID;
    float* h0    = ws;
    float* bufA  = ws + NF;
    float* bufB  = ws + 2 * NF;
    float* stats = ws + 3 * NF;
    _Float16* wf16 = (_Float16*)(ws + 3 * NF + 512);
    _Float16* fW_in   = wf16;
    _Float16* fW_l[3] = {wf16 + 1 * 16384, wf16 + 3 * 16384, wf16 + 5 * 16384};
    _Float16* fW_r[3] = {wf16 + 2 * 16384, wf16 + 4 * 16384, wf16 + 6 * 16384};
    _Float16* fW_fc1  = wf16 + 7 * 16384;
    float* h1 = bufB;   // bufB is free after layer 3's reads

    // one-time (per launch) weight conversion to f16, stays hot in L2
    cvt_f16_kernel<<<64, 256, 0, stream>>>(W_in, fW_in, 16384);
    for (int l = 0; l < 3; ++l) {
        cvt_f16_kernel<<<64, 256, 0, stream>>>(W_l[l], fW_l[l], 16384);
        cvt_f16_kernel<<<64, 256, 0, stream>>>(W_r[l], fW_r[l], 16384);
    }
    cvt_f16_kernel<<<32, 256, 0, stream>>>(W_fc1, fW_fc1, 8192);

    const int GEMM_BLOCKS = (MTILES + 3) / 4;                  // 1563
    const int EW_BLOCKS   = (N_NODES * HID / 4 + 255) / 256;   // 12500
    const int SC_BLOCKS   = (N_EDGES * 32) / 256;              // 200000

    // input projection: h0 = x @ W_in^T + b_in
    gemm_kernel<false, false, 8, false, HID><<<GEMM_BLOCKS, 128, 0, stream>>>(
        x, nullptr, fW_in, nullptr, b_in, h0, nullptr);

    const float* hcur = h0;
    float* aggbuf[3] = {bufA, bufB, bufA};
    for (int l = 0; l < 3; ++l) {
        float* agg = aggbuf[l];
        init_kernel<<<EW_BLOCKS, 256, 0, stream>>>((float4*)agg, stats);
        scatter_max_kernel<<<SC_BLOCKS, 256, 0, stream>>>(hcur, src, dst, agg);
        // in-place over agg is safe: each wave registers all its A rows first
        gemm_kernel<true, true, 8, false, HID><<<GEMM_BLOCKS, 128, 0, stream>>>(
            agg, hcur, fW_l[l], fW_r[l], b_l[l], agg, stats);
        bn_finalize_kernel<<<1, 128, 0, stream>>>(stats, g[l], be[l]);
        if (l == 2)
            bn_lrelu_kernel<true><<<EW_BLOCKS, 256, 0, stream>>>(
                (float4*)agg, stats, (const float4*)h0);
        else
            bn_lrelu_kernel<false><<<EW_BLOCKS, 256, 0, stream>>>(
                (float4*)agg, stats, nullptr);
        hcur = agg;
    }

    // fc1: 128 -> 64 with lrelu (4 N-tiles)
    gemm_kernel<false, false, 4, true, 64><<<GEMM_BLOCKS, 128, 0, stream>>>(
        hcur, nullptr, fW_fc1, nullptr, b_fc1, h1, nullptr);
    // fc2: 64 -> 1
    fc2_kernel<<<(N_NODES + 255) / 256, 256, 0, stream>>>(h1, W_fc2, b_fc2, (float*)d_out);
}
